// GraphSAGE_Nextdoor_11845519802672
// MI455X (gfx1250) — matrix operand order repaired and checked by reading, not compiled, not run on hardware
//
#include <hip/hip_runtime.h>
#include <hip/hip_bf16.h>
#include <stdint.h>

// ---------------------------------------------------------------------------
// GraphSAGE (2-layer, mean agg) for MI455X / gfx1250, wave32 + WMMA bf16.
//  - Edge pass : fused gather -> dropout -> fp32 atomic segment-sum (L2 atomics),
//                float4 (b128) coalesced feature reads
//  - Finalize  : mean + self-dropout + relu, float4, in place over sum buffer
//  - GEMMs     : v_wmma_f32_16x16x32_bf16, K fully unrolled (templated),
//                B pre-packed to padded column-major bf16 (branchless fragments)
// ---------------------------------------------------------------------------

typedef __attribute__((ext_vector_type(16))) __bf16 v16bf;
typedef __attribute__((ext_vector_type(8)))  float  v8f;

// Stateless per-(layer,row,col) dropout: keep-prob 0.5 -> scale 2.0 or 0.0.
__device__ __forceinline__ float drop_scale(uint32_t layer, uint32_t row, uint32_t col) {
    uint32_t h = (layer + 1u) * 0x9E3779B9u;
    h ^= row * 0x85EBCA6Bu + 0x27220A95u;
    h ^= col * 0xC2B2AE35u + 0x165667B1u;
    h ^= h >> 16; h *= 0x7FEB352Du;
    h ^= h >> 15; h *= 0x846CA68Bu;
    h ^= h >> 16;
    return (h & 1u) ? 2.0f : 0.0f;
}

// One wave per edge; float4 coalesced reads (d is a multiple of 128), fp32
// atomic adds into the L2-resident segment-sum buffer.
__global__ void sage_edge_agg(const int* __restrict__ src, const int* __restrict__ dst, int E,
                              const float* __restrict__ feat, const int* __restrict__ gather_idx,
                              int d, float* __restrict__ sum, float* __restrict__ deg,
                              uint32_t drop_layer) {
    int wave = (int)((blockIdx.x * (size_t)blockDim.x + threadIdx.x) >> 5);
    int lane = threadIdx.x & 31;
    if (wave >= E) return;
    int srow = src[wave];
    int drow = dst[wave];
    int node = gather_idx ? gather_idx[srow] : srow;
    const float4* fp = (const float4*)(feat + (size_t)node * d);
    float*        sp = sum + (size_t)drow * d;
    for (int base = 0; base < d; base += 128) {
        float4 v = fp[(base >> 2) + lane];
        int c = base + lane * 4;
        float m0 = v.x * drop_scale(drop_layer, (uint32_t)srow, (uint32_t)(c + 0));
        float m1 = v.y * drop_scale(drop_layer, (uint32_t)srow, (uint32_t)(c + 1));
        float m2 = v.z * drop_scale(drop_layer, (uint32_t)srow, (uint32_t)(c + 2));
        float m3 = v.w * drop_scale(drop_layer, (uint32_t)srow, (uint32_t)(c + 3));
        atomicAdd(&sp[c + 0], m0);
        atomicAdd(&sp[c + 1], m1);
        atomicAdd(&sp[c + 2], m2);
        atomicAdd(&sp[c + 3], m3);
    }
    if (lane == 0) atomicAdd(&deg[drow], 1.0f);
}

// a = relu(sum/deg (deg>0) + dropout(self)); written IN PLACE over `sum`.
// One thread per float4 (d multiple of 4).
__global__ void sage_finalize(float* __restrict__ sum, const float* __restrict__ deg,
                              const float* __restrict__ self, const int* __restrict__ self_idx,
                              int d, int n, uint32_t drop_layer) {
    size_t tid = (size_t)blockIdx.x * blockDim.x + threadIdx.x;
    int d4 = d >> 2;
    size_t total = (size_t)n * (size_t)d4;
    if (tid >= total) return;
    int row = (int)(tid / d4);
    int col = (int)(tid % d4) * 4;
    float dg = deg[row];
    float inv = (dg > 0.0f) ? (1.0f / dg) : 0.0f;
    int srow = self_idx ? self_idx[row] : row;
    float4 s  = ((const float4*)(sum  + (size_t)row  * d))[col >> 2];
    float4 h  = ((const float4*)(self + (size_t)srow * d))[col >> 2];
    float4 r;
    r.x = s.x * inv + h.x * drop_scale(drop_layer, (uint32_t)row, (uint32_t)(col + 0));
    r.y = s.y * inv + h.y * drop_scale(drop_layer, (uint32_t)row, (uint32_t)(col + 1));
    r.z = s.z * inv + h.z * drop_scale(drop_layer, (uint32_t)row, (uint32_t)(col + 2));
    r.w = s.w * inv + h.w * drop_scale(drop_layer, (uint32_t)row, (uint32_t)(col + 3));
    r.x = r.x > 0.0f ? r.x : 0.0f;
    r.y = r.y > 0.0f ? r.y : 0.0f;
    r.z = r.z > 0.0f ? r.z : 0.0f;
    r.w = r.w > 0.0f ? r.w : 0.0f;
    ((float4*)(sum + (size_t)row * d))[col >> 2] = r;
}

// Pack B[K,N] (fp32 row-major) -> Bt[Npad,K] bf16 "column-major", zero-padded
// columns. Lane B-fragments become one contiguous 32B load.
__global__ void pack_b_bf16(const float* __restrict__ B, __bf16* __restrict__ Bt,
                            int K, int N, int Npad) {
    int idx = blockIdx.x * blockDim.x + threadIdx.x;
    if (idx >= Npad * K) return;
    int col = idx / K;
    int k   = idx - col * K;
    float v = (col < N) ? B[(size_t)k * N + col] : 0.0f;
    Bt[idx] = (__bf16)v;
}

__global__ void pad_bias(const float* __restrict__ b, float* __restrict__ bp, int N, int Npad) {
    int i = blockIdx.x * blockDim.x + threadIdx.x;
    if (i < Npad) bp[i] = (i < N) ? b[i] : 0.0f;
}

// C[M,N] = A[M,K] @ B + bias.  A fp32 row-major (converted to bf16 in-reg),
// B pre-packed bf16 [Npad,K].  K templated -> fully unrolled WMMA chain.
// grid.x = ceil(M/16), grid.y = ceil(Npad/64); block = 128 (4 waves);
// wave w computes the 16x16 tile at (blockIdx.x*16, blockIdx.y*64 + 16*w).
template <int K>
__global__ void gemm_bf16_wmma(const float* __restrict__ A, const __bf16* __restrict__ Bt,
                               const float* __restrict__ biasp, float* __restrict__ C,
                               int M, int N, int Npad) {
    int wave = threadIdx.x >> 5;
    int lane = threadIdx.x & 31;
    int m0 = blockIdx.x * 16;
    int n0 = blockIdx.y * 64 + wave * 16;
    if (n0 >= Npad || m0 >= M) return;

    // A fragment: lanes 0-15 -> rows 0..15, K in {0..15} of the k-step;
    //             lanes 16-31 -> same rows,  K in {16..31}.
    int mr  = lane & 15;
    int kbA = (lane < 16) ? 0 : 16;
    // B fragment: lane = column; low lanes K 0..15, high lanes K 16..31.
    int nc  = lane & 15;
    int kbB = (lane < 16) ? 0 : 16;

    int arow = m0 + mr; if (arow >= M) arow = M - 1;       // stores are guarded
    const float*  ap = A  + (size_t)arow * K;
    int col = n0 + nc;                                     // col < Npad always
    const __bf16* bp = Bt + (size_t)col * K;

    v8f acc = {};
#pragma unroll
    for (int k0 = 0; k0 < K; k0 += 32) {
        // A: two aligned 32B groups (8 fp32 each) -> bf16
        const float4* a0 = (const float4*)(ap + k0 + kbA);
        const float4* a1 = (const float4*)(ap + k0 + kbA + 8);
        float4 x0 = a0[0], x1 = a0[1];
        float4 y0 = a1[0], y1 = a1[1];
        v16bf af;
        af[0]  = (__bf16)x0.x; af[1]  = (__bf16)x0.y; af[2]  = (__bf16)x0.z; af[3]  = (__bf16)x0.w;
        af[4]  = (__bf16)x1.x; af[5]  = (__bf16)x1.y; af[6]  = (__bf16)x1.z; af[7]  = (__bf16)x1.w;
        af[8]  = (__bf16)y0.x; af[9]  = (__bf16)y0.y; af[10] = (__bf16)y0.z; af[11] = (__bf16)y0.w;
        af[12] = (__bf16)y1.x; af[13] = (__bf16)y1.y; af[14] = (__bf16)y1.z; af[15] = (__bf16)y1.w;
        // B: one contiguous, 32B-aligned 16xbf16 vector
        v16bf bf = *(const v16bf*)(bp + k0 + kbB);
        // (neg_a, A, neg_b, B, c_mod, C, reuse_a, reuse_b)
        acc = __builtin_amdgcn_wmma_f32_16x16x32_bf16(false, af, false, bf,
                                                      (short)0, acc, false, false);
    }

    if (col < N) {
        float bv = biasp[col];
        int rowbase = m0 + ((lane >= 16) ? 8 : 0);
#pragma unroll
        for (int r = 0; r < 8; ++r) {
            int row = rowbase + r;
            if (row < M) C[(size_t)row * N + col] = acc[r] + bv;
        }
    }
}

extern "C" void kernel_launch(void* const* d_in, const int* in_sizes, int n_in,
                              void* d_out, int out_size, void* d_ws, size_t ws_size,
                              hipStream_t stream) {
    const float* features = (const float*)d_in[0];
    const int*   s0   = (const int*)d_in[1];
    const int*   s1   = (const int*)d_in[2];
    const int*   s2   = (const int*)d_in[3];
    const int*   src0 = (const int*)d_in[4];
    const int*   dst0 = (const int*)d_in[5];
    const int*   src1 = (const int*)d_in[6];
    const int*   dst1 = (const int*)d_in[7];
    const float* W1   = (const float*)d_in[8];
    const float* b1   = (const float*)d_in[9];
    const float* W2   = (const float*)d_in[10];
    const float* b2   = (const float*)d_in[11];

    const int N0 = in_sizes[1];      // 10000
    const int N1 = in_sizes[2];      // 100000
    const int E0 = in_sizes[4];      // 100000
    const int E1 = in_sizes[6];      // 1000000
    const int DIN = 128, DHID = 256, DOUT = 47, DOUTP = 48;

    // --- workspace carve-up (256B aligned) ---
    char*  ws  = (char*)d_ws;
    size_t off = 0;
    auto carve = [&](size_t bytes) -> void* {
        void* p = ws + off;
        off += (bytes + 255) & ~(size_t)255;
        return p;
    };
    size_t bigA = (size_t)N1 * DIN;                   // 12.8M floats
    size_t l2A  = (size_t)N0 * DHID;                  // 2.56M floats
    float*  bufA = (float*)carve((bigA > l2A ? bigA : l2A) * 4); // sum1/a1 -> sum2/a2
    float*  buf0 = (float*)carve((size_t)N0 * DIN * 4);          // sum0/a0
    float*  deg1 = (float*)carve((size_t)N1 * 4);
    float*  deg0 = (float*)carve((size_t)N0 * 4);
    float*  deg2 = (float*)carve((size_t)N0 * 4);
    float*  n1   = (float*)carve((size_t)N1 * DHID * 4);
    float*  n0   = (float*)carve((size_t)N0 * DHID * 4);
    __bf16* Bt1  = (__bf16*)carve((size_t)DHID * DIN * 2);       // W1 packed [256,128]
    __bf16* Bt2  = (__bf16*)carve((size_t)DOUTP * DHID * 2);     // W2 packed [48,256]
    float*  b2p  = (float*)carve((size_t)DOUTP * 4);
    (void)ws_size; (void)n_in;

    // ---- weight packing (tiny; runs every launch, deterministic) ----
    pack_b_bf16<<<(DHID * DIN + 255) / 256, 256, 0, stream>>>(W1, Bt1, DIN, DHID, DHID);
    pack_b_bf16<<<(DOUTP * DHID + 255) / 256, 256, 0, stream>>>(W2, Bt2, DHID, DOUT, DOUTP);
    pad_bias<<<1, 64, 0, stream>>>(b2, b2p, DOUT, DOUTP);

    // ---- hop 0 (small graph): a0 = agg(h1[src0]->dst0, self=h0) ----
    hipMemsetAsync(buf0, 0, (size_t)N0 * DIN * 4, stream);
    hipMemsetAsync(deg0, 0, (size_t)N0 * 4, stream);
    sage_edge_agg<<<(E0 + 7) / 8, 256, 0, stream>>>(src0, dst0, E0, features, s1,
                                                    DIN, buf0, deg0, 0u);
    {
        size_t tot = (size_t)N0 * (DIN / 4);
        sage_finalize<<<(unsigned)((tot + 255) / 256), 256, 0, stream>>>(
            buf0, deg0, features, s0, DIN, N0, 1u);
    }
    // n0 = a0 @ W1 + b1   [10000 x 256]
    {
        dim3 g((N0 + 15) / 16, DHID / 64);
        gemm_bf16_wmma<128><<<g, 128, 0, stream>>>(buf0, Bt1, b1, n0, N0, DHID, DHID);
    }

    // ---- hop 1 (big graph): a1 = agg(h2[src1]->dst1, self=h1) ----
    hipMemsetAsync(bufA, 0, (size_t)N1 * DIN * 4, stream);
    hipMemsetAsync(deg1, 0, (size_t)N1 * 4, stream);
    sage_edge_agg<<<(E1 + 7) / 8, 256, 0, stream>>>(src1, dst1, E1, features, s2,
                                                    DIN, bufA, deg1, 2u);
    {
        size_t tot = (size_t)N1 * (DIN / 4);
        sage_finalize<<<(unsigned)((tot + 255) / 256), 256, 0, stream>>>(
            bufA, deg1, features, s1, DIN, N1, 3u);
    }
    // n1 = a1 @ W1 + b1   [100000 x 256]
    {
        dim3 g((N1 + 15) / 16, DHID / 64);
        gemm_bf16_wmma<128><<<g, 128, 0, stream>>>(bufA, Bt1, b1, n1, N1, DHID, DHID);
    }

    // ---- layer 1: a = agg(n1[src0]->dst0, self=n0)  (reuses bufA region) ----
    hipMemsetAsync(bufA, 0, (size_t)N0 * DHID * 4, stream);
    hipMemsetAsync(deg2, 0, (size_t)N0 * 4, stream);
    sage_edge_agg<<<(E0 + 7) / 8, 256, 0, stream>>>(src0, dst0, E0, n1, (const int*)nullptr,
                                                    DHID, bufA, deg2, 4u);
    {
        size_t tot = (size_t)N0 * (DHID / 4);
        sage_finalize<<<(unsigned)((tot + 255) / 256), 256, 0, stream>>>(
            bufA, deg2, n0, (const int*)nullptr, DHID, N0, 5u);
    }
    // out = a @ W2 + b2   [10000 x 47]
    {
        dim3 g((N0 + 15) / 16, DOUTP / 64 ? DOUTP / 64 : 1);
        gemm_bf16_wmma<256><<<g, 128, 0, stream>>>(bufA, Bt2, b2p, (float*)d_out, N0, DOUT, DOUTP);
    }
    (void)out_size;
}
